// Speller_24086176596106
// MI455X (gfx1250) — matrix operand order repaired and checked by reading, not compiled, not run on hardware
//
#include <hip/hip_runtime.h>
#include <hip/hip_bf16.h>
#include <stddef.h>

// ---------------- problem dims ----------------
#define BS    64
#define RAL   1024
#define LAL   256
#define CS    256
#define VOC   34
#define HFS   512
#define EMBD  256
#define FOURH 2048
#define KA    1024      // concat K for both LSTM GEMMs
#define NWG   64        // persistent grid: 64 WGs x 256 threads = 512 wave32

typedef __attribute__((ext_vector_type(16))) __bf16 v16bf;
typedef __attribute__((ext_vector_type(8)))  float  v8f;

union ABf16 { uint4 u[2]; v16bf v; };
union U16x8 { uint4 u; unsigned short s[8]; };

__device__ __forceinline__ float bf2f(unsigned short h) {
  union { unsigned u; float f; } x; x.u = ((unsigned)h) << 16; return x.f;
}
__device__ __forceinline__ unsigned short f2bf(float f) {
  union { float f; unsigned u; } x; x.f = f;
  unsigned r = x.u + 0x7fffu + ((x.u >> 16) & 1u);   // RNE
  return (unsigned short)(r >> 16);
}
__device__ __forceinline__ float sigm(float x) { return 1.0f / (1.0f + __expf(-x)); }

// ---------------- device-wide sense-reversing barrier ----------------
__device__ __forceinline__ void grid_sync(unsigned* cnt, unsigned* gen, unsigned nwg) {
  __threadfence();              // make this thread's stores agent-visible
  __syncthreads();
  if (threadIdx.x == 0) {
    unsigned g0 = __hip_atomic_load(gen, __ATOMIC_RELAXED, __HIP_MEMORY_SCOPE_AGENT);
    unsigned a  = __hip_atomic_fetch_add(cnt, 1u, __ATOMIC_ACQ_REL, __HIP_MEMORY_SCOPE_AGENT);
    if (a == nwg - 1u) {
      __hip_atomic_store(cnt, 0u, __ATOMIC_RELAXED, __HIP_MEMORY_SCOPE_AGENT);
      __hip_atomic_fetch_add(gen, 1u, __ATOMIC_RELEASE, __HIP_MEMORY_SCOPE_AGENT);
    } else {
      while (__hip_atomic_load(gen, __ATOMIC_ACQUIRE, __HIP_MEMORY_SCOPE_AGENT) == g0)
        __builtin_amdgcn_s_sleep(2);
    }
  }
  __syncthreads();
}

// ---------------- fused LSTM GEMM (bf16 WMMA) + gates ----------------
// X  : [64][1024] bf16 (A matrix, row-major)
// W  : [2048][1024] bf16, rows permuted so dest row = h*4+gate  (gate = i,f,g,o)
// One 16x16 tile per wave; K = 1024 -> 32 x v_wmma_f32_16x16x32_bf16.
// Gate nonlinearity fused in epilogue via intra-quad shuffles.
__device__ __forceinline__ void lstm_phase(
    const unsigned short* __restrict__ X,
    const unsigned short* __restrict__ W,
    const float* __restrict__ bias,
    float* __restrict__ cst,                 // [64][512] f32 cell state (RMW)
    unsigned short* __restrict__ h0,         // h dest 0: index [m*1024 + h]
    unsigned short* __restrict__ h1,         // optional h dest 1 (or null)
    unsigned char* smem, int g, int tid)
{
  const int lane = tid & 31, j = tid >> 5;
  const int mt = g >> 4;                     // 4 row-tiles, 16 WGs each
  const int nt = ((g & 15) << 3) + j;        // 128 col-tiles
  // ---- stage A rows (16 x 1024 bf16 = 32KB) into LDS ----
  const uint4* src = (const uint4*)(X + (size_t)mt * 16 * KA);
  uint4* sA = (uint4*)smem;
  for (int i = tid; i < 2048; i += 256) sA[i] = src[i];
  __syncthreads();

  const int m = lane & 15, grp = lane >> 4;
  float bv = bias[nt * 16 + m];
  v8f acc = { bv, bv, bv, bv, bv, bv, bv, bv };

  const uint4* Bp = (const uint4*)W;
  const size_t arow = (size_t)m * 128 + grp * 2;             // LDS uint4 units
  const size_t brow = (size_t)(nt * 16 + m) * 128 + grp * 2; // global uint4 units

  for (int kk = 0; kk < 32; ++kk) {
    ABf16 a, b;
    a.u[0] = sA[arow + kk * 4];
    a.u[1] = sA[arow + kk * 4 + 1];
    b.u[0] = Bp[brow + kk * 4];
    b.u[1] = Bp[brow + kk * 4 + 1];
    if (kk < 28) __builtin_prefetch((const void*)(Bp + brow + (kk + 4) * 4), 0, 3);
    acc = __builtin_amdgcn_wmma_f32_16x16x32_bf16(false, a.v, false, b.v,
                                                  (short)0, acc, false, false);
  }

  // ---- epilogue: tile columns hold (i,f,g,o) of 4 hidden units ----
  const int qlane = lane & 3;
  const int hl = nt * 4 + ((lane & 15) >> 2);
  const int base = lane & ~3;
#pragma unroll
  for (int r = 0; r < 8; ++r) {
    float z  = acc[r];
    float zi = __shfl(z, base + 0, 32);
    float zf = __shfl(z, base + 1, 32);
    float zg = __shfl(z, base + 2, 32);
    float zo = __shfl(z, base + 3, 32);
    if (qlane == 0) {
      int mrow = mt * 16 + ((lane < 16) ? r : r + 8);
      float c_old = cst[mrow * HFS + hl];
      float cn = sigm(zf) * c_old + sigm(zi) * tanhf(zg);
      float hn = sigm(zo) * tanhf(cn);
      cst[mrow * HFS + hl] = cn;
      unsigned short hb = f2bf(hn);
      h0[mrow * KA + hl] = hb;
      if (h1) h1[mrow * KA + hl] = hb;
    }
  }
}

// ---------------- persistent main kernel ----------------
struct Params {
  const unsigned short *Wa, *Wb, *Wq;
  const float *ba, *bb, *bq, *Wo, *bo, *emb;
  const unsigned short *key, *valt;    // valt: [B][C][R] transposed bf16
  const unsigned char* mask;
  const int* y;
  unsigned short *xa0, *xa1, *xb0, *xb1, *qbf;
  float *ca, *cb, *out;
  unsigned *bcnt, *bgen;
};

__global__ void __launch_bounds__(256, 1) speller_main(Params p) {
  __shared__ alignas(16) unsigned char smem[32768];
  const int tid = threadIdx.x, g = blockIdx.x;
  const int lane = tid & 31, j = tid >> 5;

  for (int t = 0; t < LAL; ++t) {
    unsigned short* xaR = (t & 1) ? p.xa1 : p.xa0;   // read at step t
    unsigned short* xaW = (t & 1) ? p.xa0 : p.xa1;   // written for step t+1
    unsigned short* xbR = (t & 1) ? p.xb1 : p.xb0;
    unsigned short* xbW = (t & 1) ? p.xb0 : p.xb1;

    // ---- P1: LSTM A.  z = [e,ctx,ha] @ Wa^T  -> ha(t), ca(t) ----
    lstm_phase(xaR, p.Wa, p.ba, p.ca, xaW + HFS, xbR, smem, g, tid);
    grid_sync(p.bcnt, p.bgen, NWG);

    // ---- P2: LSTM B.  z = [ha,hb] @ Wb^T -> hb(t), cb(t) ----
    lstm_phase(xbR, p.Wb, p.bb, p.cb, xbW + HFS, nullptr, smem, g, tid);
    grid_sync(p.bcnt, p.bgen, NWG);

    // ---- P3: query = hb @ Wq^T (+ emb gather for t+1 on spare WGs) ----
    if (g < 8) {
      const int tile = g * 8 + j, mt = tile >> 4, nt = tile & 15;
      const uint4* gx = (const uint4*)xbW;
      uint4* sA = (uint4*)smem;
      for (int i = tid; i < 1024; i += 256) {      // 16 rows x 512 bf16 of hb
        int r = i >> 6, cu = i & 63;
        sA[i] = gx[(size_t)(mt * 16 + r) * 128 + 64 + cu];
      }
      __syncthreads();
      const int m = lane & 15, grp = lane >> 4;
      float bv = p.bq[nt * 16 + m];
      v8f acc = { bv, bv, bv, bv, bv, bv, bv, bv };
      const uint4* Bq = (const uint4*)p.Wq;
      const size_t arow = (size_t)m * 64 + grp * 2;
      const size_t brow = (size_t)(nt * 16 + m) * 64 + grp * 2;
      for (int kk = 0; kk < 16; ++kk) {
        ABf16 a, b;
        a.u[0] = sA[arow + kk * 4];
        a.u[1] = sA[arow + kk * 4 + 1];
        b.u[0] = Bq[brow + kk * 4];
        b.u[1] = Bq[brow + kk * 4 + 1];
        acc = __builtin_amdgcn_wmma_f32_16x16x32_bf16(false, a.v, false, b.v,
                                                      (short)0, acc, false, false);
      }
#pragma unroll
      for (int r = 0; r < 8; ++r) {
        int mrow = mt * 16 + ((lane < 16) ? r : r + 8);
        p.qbf[mrow * CS + nt * 16 + m] = f2bf(acc[r]);
      }
    } else if (g < 16 && t < LAL - 1) {
      int f = (g - 8) * 256 + tid;                  // gather e(t+1) -> xaW[:,0:256]
      for (int idx = f; idx < BS * EMBD; idx += 2048) {
        int b = idx >> 8, c = idx & 255;
        int tok = p.y[b * LAL + (t + 1)];
        xaW[b * KA + c] = f2bf(p.emb[tok * EMBD + c]);
      }
    }
    grid_sync(p.bcnt, p.bgen, NWG);

    // ---- P4: attention (one batch per WG) + output projection ----
    {
      const int b = g;
      float* qf   = (float*)smem;        // 256
      float* pr   = qf + 256;            // 1024 (16B aligned)
      float* red  = pr + 1024;           // 256
      float* ctxf = red + 256;           // 256
      qf[tid] = bf2f(p.qbf[b * CS + tid]);
      __syncthreads();

      float e[4];
#pragma unroll
      for (int k = 0; k < 4; ++k) {
        int r = tid + k * 256;
        const uint4* kp = (const uint4*)(p.key + ((size_t)b * RAL + r) * CS);
        float s = 0.f;
        for (int cc = 0; cc < 32; ++cc) {
          U16x8 kk; kk.u = kp[cc];
#pragma unroll
          for (int jj = 0; jj < 8; ++jj) s += qf[cc * 8 + jj] * bf2f(kk.s[jj]);
        }
        if (!p.mask[b * RAL + r]) s = -1e9f;
        e[k] = s;
      }
      red[tid] = fmaxf(fmaxf(e[0], e[1]), fmaxf(e[2], e[3]));
      __syncthreads();
      for (int s = 128; s > 0; s >>= 1) {
        if (tid < s) red[tid] = fmaxf(red[tid], red[tid + s]);
        __syncthreads();
      }
      float mx = red[0];
      __syncthreads();
      float ls = 0.f;
#pragma unroll
      for (int k = 0; k < 4; ++k) {
        float pe = __expf(e[k] - mx);
        pr[tid + k * 256] = pe;
        ls += pe;
      }
      red[tid] = ls;
      __syncthreads();
      for (int s = 128; s > 0; s >>= 1) {
        if (tid < s) red[tid] += red[tid + s];
        __syncthreads();
      }
      float rinv = 1.f / red[0];
      __syncthreads();

      // ctx[c] = sum_r attn[r]*val[b][r][c]; val is transposed so this thread
      // streams a contiguous bf16 row with b128 loads; p via LDS float4 loads.
      const uint4*  vp = (const uint4*)(p.valt + ((size_t)b * CS + tid) * RAL);
      const float4* pp = (const float4*)pr;
      float a0 = 0.f;
#pragma unroll 2
      for (int rq = 0; rq < RAL / 8; ++rq) {
        U16x8 vv; vv.u = vp[rq];
        float4 pa = pp[2 * rq], pb = pp[2 * rq + 1];
        a0 += pa.x * bf2f(vv.s[0]);
        a0 += pa.y * bf2f(vv.s[1]);
        a0 += pa.z * bf2f(vv.s[2]);
        a0 += pa.w * bf2f(vv.s[3]);
        a0 += pb.x * bf2f(vv.s[4]);
        a0 += pb.y * bf2f(vv.s[5]);
        a0 += pb.z * bf2f(vv.s[6]);
        a0 += pb.w * bf2f(vv.s[7]);
      }
      float cx = a0 * rinv;
      ctxf[tid] = cx;
      xaW[b * KA + CS + tid] = f2bf(cx);
      __syncthreads();

      // parallel output projection: 34 vocab x 7 slices of 112 elems
      {
        int v = tid / 7, part = tid % 7;
        if (v < VOC) {
          const unsigned short* hb = xbW + (size_t)b * KA + HFS;
          const float* wo = p.Wo + v * (HFS + CS);
          int k0 = part * 112;
          int k1 = (k0 + 112 < HFS + CS) ? (k0 + 112) : (HFS + CS);
          float s = 0.f;
          for (int k = k0; k < k1; ++k) {
            float xv = (k < HFS) ? bf2f(hb[k]) : ctxf[k - HFS];
            s += xv * wo[k];
          }
          red[v * 7 + part] = s;
        }
      }
      __syncthreads();
      if (tid < VOC) {
        float o = p.bo[tid];
#pragma unroll
        for (int q2 = 0; q2 < 7; ++q2) o += red[tid * 7 + q2];
        p.out[((size_t)b * LAL + t) * VOC + tid] = o;
      }
    }
    grid_sync(p.bcnt, p.bgen, NWG);
  }
}

// ---------------- prep kernels ----------------
__global__ void prep_zero(unsigned* dst, size_t n) {
  for (size_t i = blockIdx.x * (size_t)blockDim.x + threadIdx.x; i < n;
       i += (size_t)gridDim.x * blockDim.x) dst[i] = 0u;
}

__global__ void prep_weights(const float* Wih_a, const float* Whh_a,
                             const float* bih_a, const float* bhh_a,
                             const float* Wih_b, const float* Whh_b,
                             const float* bih_b, const float* bhh_b,
                             const float* Wq,
                             unsigned short* Wa, unsigned short* Wb,
                             unsigned short* Wqb, float* ba, float* bb) {
  const size_t NW = (size_t)FOURH * KA;
  const size_t NQ = (size_t)CS * HFS;
  const size_t total = 2 * NW + NQ + 2 * FOURH;
  for (size_t i = blockIdx.x * (size_t)blockDim.x + threadIdx.x; i < total;
       i += (size_t)gridDim.x * blockDim.x) {
    if (i < NW) {                                    // Wa: gate-interleaved rows
      size_t d = i / KA, k = i % KA, h = d >> 2, q = d & 3, s = q * HFS + h;
      float v = (k < HFS) ? Wih_a[s * HFS + k] : Whh_a[s * HFS + (k - HFS)];
      Wa[i] = f2bf(v);
    } else if (i < 2 * NW) {
      size_t jx = i - NW, d = jx / KA, k = jx % KA, h = d >> 2, q = d & 3, s = q * HFS + h;
      float v = (k < HFS) ? Wih_b[s * HFS + k] : Whh_b[s * HFS + (k - HFS)];
      Wb[jx] = f2bf(v);
    } else if (i < 2 * NW + NQ) {
      size_t jx = i - 2 * NW;
      Wqb[jx] = f2bf(Wq[jx]);
    } else if (i < 2 * NW + NQ + FOURH) {
      size_t d = i - 2 * NW - NQ, h = d >> 2, q = d & 3, s = q * HFS + h;
      ba[d] = bih_a[s] + bhh_a[s];
    } else {
      size_t d = i - 2 * NW - NQ - FOURH, h = d >> 2, q = d & 3, s = q * HFS + h;
      bb[d] = bih_b[s] + bhh_b[s];
    }
  }
}

// key: straight f32->bf16; val: transpose per batch to [B][C][R] bf16
__global__ void prep_kv(const float* key, const float* val,
                        unsigned short* kb, unsigned short* vt) {
  const size_t N = (size_t)BS * RAL * CS;
  for (size_t i = blockIdx.x * (size_t)blockDim.x + threadIdx.x; i < 2 * N;
       i += (size_t)gridDim.x * blockDim.x) {
    if (i < N) {
      kb[i] = f2bf(key[i]);
    } else {
      size_t jx = i - N;
      size_t b = jx / ((size_t)RAL * CS);
      size_t rem = jx % ((size_t)RAL * CS);
      size_t r = rem / CS, c = rem % CS;
      vt[b * (size_t)CS * RAL + c * RAL + r] = f2bf(val[jx]);
    }
  }
}

__global__ void prep_embed0(const float* emb, const int* y, unsigned short* xa0) {
  int b = blockIdx.x, c = threadIdx.x;
  int tok = y[b * LAL + 0];
  xa0[b * KA + c] = f2bf(emb[tok * EMBD + c]);
}

// ---------------- host launch ----------------
extern "C" void kernel_launch(void* const* d_in, const int* in_sizes, int n_in,
                              void* d_out, int out_size, void* d_ws, size_t ws_size,
                              hipStream_t stream) {
  const float* key   = (const float*)d_in[0];
  const float* val   = (const float*)d_in[1];
  const int*   y     = (const int*)d_in[2];
  const unsigned char* mask = (const unsigned char*)d_in[3];
  const float* emb   = (const float*)d_in[4];
  const float* Wih_a = (const float*)d_in[5];
  const float* Whh_a = (const float*)d_in[6];
  const float* bih_a = (const float*)d_in[7];
  const float* bhh_a = (const float*)d_in[8];
  const float* Wih_b = (const float*)d_in[9];
  const float* Whh_b = (const float*)d_in[10];
  const float* bih_b = (const float*)d_in[11];
  const float* bhh_b = (const float*)d_in[12];
  const float* Wq    = (const float*)d_in[13];
  const float* bq    = (const float*)d_in[14];
  const float* Wo    = (const float*)d_in[15];
  const float* bo    = (const float*)d_in[16];

  unsigned char* ws = (unsigned char*)d_ws;
  size_t off = 0;
  auto take = [&](size_t n) { size_t o = off; off += (n + 255) & ~(size_t)255; return o; };
  size_t oWa  = take((size_t)FOURH * KA * 2);
  size_t oWb  = take((size_t)FOURH * KA * 2);
  size_t oWq  = take((size_t)CS * HFS * 2);
  size_t oBa  = take((size_t)FOURH * 4);
  size_t oBb  = take((size_t)FOURH * 4);
  size_t oKey = take((size_t)BS * RAL * CS * 2);
  size_t oVal = take((size_t)BS * RAL * CS * 2);
  size_t oXa  = take((size_t)2 * BS * KA * 2);     // state region starts here
  size_t oXb  = take((size_t)2 * BS * KA * 2);
  size_t oCa  = take((size_t)BS * HFS * 4);
  size_t oCb  = take((size_t)BS * HFS * 4);
  size_t oQb  = take((size_t)BS * CS * 2);
  size_t oBar = take(256);
  // total ~77MB; ws_size must cover it

  unsigned short* Wa  = (unsigned short*)(ws + oWa);
  unsigned short* Wb  = (unsigned short*)(ws + oWb);
  unsigned short* Wqb = (unsigned short*)(ws + oWq);
  float* ba = (float*)(ws + oBa);
  float* bb = (float*)(ws + oBb);
  unsigned short* kb  = (unsigned short*)(ws + oKey);
  unsigned short* vt  = (unsigned short*)(ws + oVal);
  unsigned short* xa0 = (unsigned short*)(ws + oXa);
  unsigned short* xa1 = xa0 + (size_t)BS * KA;
  unsigned short* xb0 = (unsigned short*)(ws + oXb);
  unsigned short* xb1 = xb0 + (size_t)BS * KA;
  float* ca = (float*)(ws + oCa);
  float* cb = (float*)(ws + oCb);
  unsigned short* qbf = (unsigned short*)(ws + oQb);
  unsigned* bar = (unsigned*)(ws + oBar);

  // zero all state (xa/xb/ca/cb/qbf/barrier) — ws is poisoned, must init each call
  size_t zero_words = (off - oXa) / 4;
  prep_zero<<<512, 256, 0, stream>>>((unsigned*)(ws + oXa), zero_words);
  prep_weights<<<2048, 256, 0, stream>>>(Wih_a, Whh_a, bih_a, bhh_a,
                                         Wih_b, Whh_b, bih_b, bhh_b,
                                         Wq, Wa, Wb, Wqb, ba, bb);
  prep_kv<<<4096, 256, 0, stream>>>(key, val, kb, vt);
  prep_embed0<<<BS, EMBD, 0, stream>>>(emb, y, xa0);

  Params P;
  P.Wa = Wa; P.Wb = Wb; P.Wq = Wqb;
  P.ba = ba; P.bb = bb; P.bq = bq; P.Wo = Wo; P.bo = bo; P.emb = emb;
  P.key = kb; P.valt = vt; P.mask = mask; P.y = y;
  P.xa0 = xa0; P.xa1 = xa1; P.xb0 = xb0; P.xb1 = xb1; P.qbf = qbf;
  P.ca = ca; P.cb = cb; P.out = (float*)d_out;
  P.bcnt = bar; P.bgen = bar + 1;

  speller_main<<<NWG, 256, 0, stream>>>(P);
}
